// SNN_Delay_29171417874638
// MI455X (gfx1250) — matrix operand order repaired and checked by reading, not compiled, not run on hardware
//
#include <hip/hip_runtime.h>
#include <hip/hip_bf16.h>

// ---------------------------------------------------------------------------
// SNN with learnable delays (DCLS gauss) on MI455X / gfx1250.
// Delay-conv == sum of 5 shifted GEMMs on the WMMA pipe
// (v_wmma_f32_16x16x32_f16, fp32 accumulation). LIF scans stay VALU.
// This revision: ping-pong (unroll-by-2) fragment buffers -> zero register
// moves in the steady state, loads consumed ~8 WMMAs after issue, staggered
// s_wait_loadcnt instead of wait(0) + dual_mov swap storm.
// ---------------------------------------------------------------------------

typedef __attribute__((ext_vector_type(16))) _Float16 v16h;
typedef __attribute__((ext_vector_type(8)))  _Float16 v8h;
typedef __attribute__((ext_vector_type(8)))  float    v8f;

// Problem constants (from reference)
#define BATCH   128
#define TSTEPS  50
#define CIN     2312
#define HID     300
#define NOUT    10
#define KS      5
#define LPAD    5
#define T1      51      // TSTEPS + LPAD - KS + 1
#define T2      52      // T1 + LPAD - KS + 1
#define BETA    0.9f
#define THRESH  1.0f
#define SIG     2.0f

// Padded GEMM dims
#define CINP    2336    // CIN padded to 32  -> 73 K-tiles
#define KT1     73
#define NT1TOT  20      // HID padded to 320 -> 20 N-tiles (5 groups of 4)
#define NTG1    4       // N-tiles per wave, layer 1
#define NGRP1   5
#define HIDP2   320     // HID padded to 32 for layer-2 K dim -> 10 K-tiles
#define KT2     10
#define NT2TOT  1       // NOUT padded to 16 -> 1 N-tile

// ---------------------------------------------------------------------------
// 1) Convert input activations fp32 -> f16, zero-pad CIN to CINP.
//    data: [B, T, CIN] -> dataH: [B, T, CINP]
// ---------------------------------------------------------------------------
__global__ __launch_bounds__(256) void cvt_data_kernel(
    const float* __restrict__ data, _Float16* __restrict__ out, long total) {
  long id = (long)blockIdx.x * blockDim.x + threadIdx.x;
  if (id >= total) return;
  int  i  = (int)(id % CINP);
  long bt = id / CINP;
  _Float16 v = (_Float16)0.f;
  if (i < CIN) v = (_Float16)data[bt * CIN + i];
  out[id] = v;
}

// ---------------------------------------------------------------------------
// 2) Synthesize DCLS gauss kernel and pack directly into WMMA B-fragment
//    order: Bpk[((k*ktiles + kt)*ntiles + nt)*512 + lane*16 + e], f16.
//    B-matrix (32x16 f16) lane layout: N = lane&15, K = (lane>>4)*8 + (e<8 ? e : e+8).
// ---------------------------------------------------------------------------
__global__ __launch_bounds__(256) void pack_weights_kernel(
    const float* __restrict__ W, const float* __restrict__ P,
    _Float16* __restrict__ Bpk, int Cin, int ktiles, int ntiles, int Nout,
    long total) {
  long id = (long)blockIdx.x * blockDim.x + threadIdx.x;
  if (id >= total) return;
  int e    = (int)(id & 15);
  int lane = (int)((id >> 4) & 31);
  long r   = id >> 9;
  int nt   = (int)(r % ntiles); r /= ntiles;
  int kt   = (int)(r % ktiles);
  int k    = (int)(r / ktiles);
  int klocal = ((lane >> 4) * 8) + (e < 8 ? e : e + 8);
  int i = kt * 32 + klocal;
  int n = nt * 16 + (lane & 15);
  _Float16 v = (_Float16)0.f;
  if (i < Cin && n < Nout) {
    float c = P[(size_t)n * Cin + i] + (float)(KS / 2);   // center = P + 2
    float gsum = 0.f;
#pragma unroll
    for (int j = 0; j < KS; ++j) {
      float d = ((float)j - c) * (1.0f / SIG);
      gsum += __expf(-0.5f * d * d);
    }
    float dk = ((float)k - c) * (1.0f / SIG);
    float g  = __expf(-0.5f * dk * dk) / (gsum + 1e-7f);
    v = (_Float16)(W[(size_t)n * Cin + i] * g);
  }
  Bpk[id] = v;
}

// ---------------------------------------------------------------------------
// Fragment helpers for the WMMA GEMM.
// ---------------------------------------------------------------------------
template <int NTG>
__device__ __forceinline__ void load_frags(
    const _Float16* __restrict__ rowPtr, const _Float16* __restrict__ bPtr,
    size_t bStride, int kt, v8h& lo, v8h& hi, v16h* bf) {
  const _Float16* r = rowPtr + (size_t)kt * 32;
  lo = *(const v8h*)(r);
  hi = *(const v8h*)(r + 16);
  const _Float16* bb = bPtr + (size_t)kt * bStride;
#pragma unroll
  for (int nt = 0; nt < NTG; ++nt)
    bf[nt] = *(const v16h*)(bb + (size_t)nt * 512);
}

template <int NTG>
__device__ __forceinline__ void mma_frags(
    const v8h& lo, const v8h& hi, const v16h* bf, v8f* acc) {
  v16h afrag;
#pragma unroll
  for (int e = 0; e < 8; ++e) { afrag[e] = lo[e]; afrag[e + 8] = hi[e]; }
#pragma unroll
  for (int nt = 0; nt < NTG; ++nt)
    acc[nt] = __builtin_amdgcn_wmma_f32_16x16x32_f16(
        false, afrag, false, bf[nt], (short)0, acc[nt], false, false);
}

// ---------------------------------------------------------------------------
// 3) Delay-conv GEMM on the WMMA pipe, ping-pong pipelined.
//    M = Tout*128 (m = t*128 + b), K = taps * Kpad, N = ntilesTot*16.
//    blockIdx.x = output time step t (8 waves = 8 b-blocks of 16 rows),
//    blockIdx.y = N-group (NTG tiles per wave).
//    A (16x32 f16) lane layout: M = lane&15,
//      K = (lane>>4)*8 + (e<8 ? e : e+8)  -> two contiguous 16B runs per lane.
//    Y layout: [Tout, 128, Nout] fp32 (+bias), for coalesced LIF reads.
//    Requires ktiles >= 2 (we use 73 and 10).
// ---------------------------------------------------------------------------
template <int NTG>
__global__ __launch_bounds__(256) void snn_gemm_wmma(
    const _Float16* __restrict__ A, int Ttot, int Kpad,
    const _Float16* __restrict__ Bpk, int ktiles, int ntilesTot,
    const float* __restrict__ bias,
    float* __restrict__ Y, int Nout) {
  const int t        = blockIdx.x;            // output time step
  const int ntBase   = blockIdx.y * NTG;      // first N-tile for this wave
  const int warp     = threadIdx.x >> 5;      // b-block 0..7
  const int lane     = threadIdx.x & 31;
  const int laneM    = lane & 15;
  const int laneHalf = lane >> 4;
  const int b        = warp * 16 + laneM;     // this lane's A row (batch idx)

  v8f acc[NTG];
#pragma unroll
  for (int nt = 0; nt < NTG; ++nt)
#pragma unroll
    for (int q = 0; q < 8; ++q) acc[nt][q] = 0.f;

  const size_t bStride = (size_t)ntilesTot * 512;   // f16 elems per kt slab

#pragma unroll
  for (int k = 0; k < KS; ++k) {
    const int tau = t + k - LPAD;             // uniform across the wave
    if (tau < 0) continue;                    // left zero-padding tap
    const _Float16* rowPtr =
        A + ((size_t)b * Ttot + tau) * (size_t)Kpad + laneHalf * 8;
    const _Float16* bPtr =
        Bpk + ((size_t)(k * ktiles) * ntilesTot + ntBase) * 512 +
        (size_t)lane * 16;

    // Ping-pong fragment buffers: buffer roles alternate -> no swap moves.
    v8h a0l, a0h, a1l, a1h;
    v16h b0[NTG], b1[NTG];
    load_frags<NTG>(rowPtr, bPtr, bStride, 0, a0l, a0h, b0);
    load_frags<NTG>(rowPtr, bPtr, bStride, 1, a1l, a1h, b1);

    int kt = 0;
    for (; kt + 3 < ktiles; kt += 2) {
      mma_frags<NTG>(a0l, a0h, b0, acc);
      load_frags<NTG>(rowPtr, bPtr, bStride, kt + 2, a0l, a0h, b0);
      mma_frags<NTG>(a1l, a1h, b1, acc);
      load_frags<NTG>(rowPtr, bPtr, bStride, kt + 3, a1l, a1h, b1);
    }
    // Tail: ktiles - kt is 2 or 3; both buffers hold valid fragments.
    mma_frags<NTG>(a0l, a0h, b0, acc);                      // frag kt
    if (ktiles - kt == 3) {
      load_frags<NTG>(rowPtr, bPtr, bStride, kt + 2, a0l, a0h, b0);
      mma_frags<NTG>(a1l, a1h, b1, acc);                    // frag kt+1
      mma_frags<NTG>(a0l, a0h, b0, acc);                    // frag kt+2
    } else {
      mma_frags<NTG>(a1l, a1h, b1, acc);                    // frag kt+1
    }
  }

  // Store C tiles: VGPR q holds (M = q + laneHalf*8, N = laneM)
#pragma unroll
  for (int nt = 0; nt < NTG; ++nt) {
    int n = (ntBase + nt) * 16 + laneM;
    if (n < Nout) {
      float bv = bias[n];
#pragma unroll
      for (int q = 0; q < 8; ++q) {
        int bo = warp * 16 + q + laneHalf * 8;
        Y[((size_t)t * BATCH + bo) * (size_t)Nout + n] = acc[nt][q] + bv;
      }
    }
  }
}

// ---------------------------------------------------------------------------
// 4) LIF scan for layer 1. 128*320 threads; each walks its own recurrence.
//    Writes spikes as f16 in [B, T1, HIDP2] (pre-padded for layer-2 GEMM A).
// ---------------------------------------------------------------------------
__global__ __launch_bounds__(256) void lif1_kernel(
    const float* __restrict__ y1, _Float16* __restrict__ spk1) {
  int id = blockIdx.x * blockDim.x + threadIdx.x;
  if (id >= BATCH * HIDP2) return;
  int bb = id / HIDP2;
  int n  = id % HIDP2;
  float mem = 0.f;
  for (int t = 0; t < T1; ++t) {
    float s = 0.f;
    if (n < HID) {
      float x = y1[((size_t)t * BATCH + bb) * HID + n];
      float reset = (mem > THRESH) ? THRESH : 0.f;
      mem = BETA * mem + x - reset;
      s = (mem > THRESH) ? 1.f : 0.f;
    }
    spk1[((size_t)bb * T1 + t) * HIDP2 + n] = (_Float16)s;
  }
}

// ---------------------------------------------------------------------------
// 5) LIF scan for layer 2 -> final outputs (spk2 then mem2, each [T2,B,NOUT]).
// ---------------------------------------------------------------------------
__global__ __launch_bounds__(256) void lif2_kernel(
    const float* __restrict__ y2, float* __restrict__ out) {
  int id = blockIdx.x * blockDim.x + threadIdx.x;
  if (id >= BATCH * NOUT) return;
  int bb = id / NOUT;
  int o  = id % NOUT;
  const size_t OFF = (size_t)T2 * BATCH * NOUT;
  float mem = 0.f;
  for (int t = 0; t < T2; ++t) {
    float x = y2[((size_t)t * BATCH + bb) * NOUT + o];
    float reset = (mem > THRESH) ? THRESH : 0.f;
    mem = BETA * mem + x - reset;
    float s = (mem > THRESH) ? 1.f : 0.f;
    size_t idx = (size_t)t * BATCH * NOUT + (size_t)bb * NOUT + o;
    out[idx]       = s;
    out[OFF + idx] = mem;
  }
}

// ---------------------------------------------------------------------------
// Launch: inputs = {data, W1, P1, b1, W2, P2, b2}, all fp32.
// Workspace layout (bytes, all 256-aligned):
//   dataH  f16 [128,50,2336]          29,900,800
//   B1pk   f16 packed 5*73*20*512      7,475,200
//   B2pk   f16 packed 5*10*1*512          51,200
//   y1     f32 [51,128,300]            7,833,600
//   spk1   f16 [128,51,320]            4,177,920
//   y2     f32 [52,128,10]               266,240
//   total ~= 49.7 MB
// ---------------------------------------------------------------------------
extern "C" void kernel_launch(void* const* d_in, const int* in_sizes, int n_in,
                              void* d_out, int out_size, void* d_ws, size_t ws_size,
                              hipStream_t stream) {
  const float* data = (const float*)d_in[0];
  const float* W1   = (const float*)d_in[1];
  const float* P1   = (const float*)d_in[2];
  const float* b1   = (const float*)d_in[3];
  const float* W2   = (const float*)d_in[4];
  const float* P2   = (const float*)d_in[5];
  const float* b2   = (const float*)d_in[6];
  float* out = (float*)d_out;

  char* w = (char*)d_ws;
  const size_t SZ_DATAH = (size_t)BATCH * TSTEPS * CINP * 2;     // 29,900,800
  const size_t SZ_B1PK  = (size_t)KS * KT1 * NT1TOT * 512 * 2;   //  7,475,200
  const size_t SZ_B2PK  = (size_t)KS * KT2 * NT2TOT * 512 * 2;   //     51,200
  const size_t SZ_Y1    = (size_t)T1 * BATCH * HID * 4;          //  7,833,600
  const size_t SZ_SPK1  = (size_t)BATCH * T1 * HIDP2 * 2;        //  4,177,920

  _Float16* dataH = (_Float16*)(w);
  _Float16* B1pk  = (_Float16*)(w + SZ_DATAH);
  _Float16* B2pk  = (_Float16*)(w + SZ_DATAH + SZ_B1PK);
  float*    y1    = (float*)   (w + SZ_DATAH + SZ_B1PK + SZ_B2PK);
  _Float16* spk1  = (_Float16*)(w + SZ_DATAH + SZ_B1PK + SZ_B2PK + SZ_Y1);
  float*    y2    = (float*)   (w + SZ_DATAH + SZ_B1PK + SZ_B2PK + SZ_Y1 + SZ_SPK1);

  // 1) cast+pad input activations
  {
    long total = (long)BATCH * TSTEPS * CINP;
    int blocks = (int)((total + 255) / 256);
    cvt_data_kernel<<<blocks, 256, 0, stream>>>(data, dataH, total);
  }
  // 2) synthesize + fragment-pack both weight tensors
  {
    long total = (long)KS * KT1 * NT1TOT * 512;
    int blocks = (int)((total + 255) / 256);
    pack_weights_kernel<<<blocks, 256, 0, stream>>>(W1, P1, B1pk, CIN, KT1, NT1TOT, HID, total);
  }
  {
    long total = (long)KS * KT2 * NT2TOT * 512;
    int blocks = (int)((total + 255) / 256);
    pack_weights_kernel<<<blocks, 256, 0, stream>>>(W2, P2, B2pk, HID, KT2, NT2TOT, NOUT, total);
  }
  // 3) layer-1 delay-conv GEMM on the WMMA pipe (45.3 GFLOP, 255 blocks)
  snn_gemm_wmma<NTG1><<<dim3(T1, NGRP1), 256, 0, stream>>>(
      dataH, TSTEPS, CINP, B1pk, KT1, NT1TOT, b1, y1, HID);
  // 4) LIF scan 1 -> f16 packed spikes
  {
    int total = BATCH * HIDP2;
    lif1_kernel<<<(total + 255) / 256, 256, 0, stream>>>(y1, spk1);
  }
  // 5) layer-2 delay-conv GEMM
  snn_gemm_wmma<NT2TOT><<<dim3(T2, 1), 256, 0, stream>>>(
      spk1, T1, HIDP2, B2pk, KT2, NT2TOT, b2, y2, NOUT);
  // 6) LIF scan 2 -> (spk2, mem2) into d_out
  {
    int total = BATCH * NOUT;
    lif2_kernel<<<(total + 255) / 256, 256, 0, stream>>>(y2, out);
  }
}
